// Mamba2LM_62053687492932
// MI455X (gfx1250) — compile-verified, matrix operand order checked by reading, not run on hardware
//
#include <hip/hip_runtime.h>
#include <hip/hip_bf16.h>
#include <stdint.h>

// ---------------- model constants ----------------
#define BATCH    2
#define SEQ      1024
#define NTOK     (BATCH * SEQ)          // 2048
#define DIM      1024
#define EXPAND   2
#define D_INNER  (EXPAND * DIM)         // 2048
#define D_STATE  128
#define HEADDIM  64
#define NHEADS   (D_INNER / HEADDIM)    // 32
#define D_CONV   4
#define CONV_DIM (D_INNER + 2 * D_STATE)            // 2304
#define D_IN_PROJ (2 * D_INNER + 2 * D_STATE + NHEADS) // 4384
#define VOCAB    32000
#define DEPTH    2

typedef __bf16 bf16_t;
typedef __attribute__((ext_vector_type(16))) __bf16 v16bf;
typedef __attribute__((ext_vector_type(8)))  float  v8f;

// ---------------- helpers ----------------
__device__ __forceinline__ float silu_f(float v) {
    return v / (1.0f + __expf(-v));
}

// CDNA5 async copy: 16B global -> LDS, tracked by ASYNCcnt (ISA 15.18.3 op 98)
__device__ __forceinline__ void async_copy_b128(uint32_t lds_addr,
                                                const void* gptr) {
    asm volatile("global_load_async_to_lds_b128 %0, %1, off"
                 :: "v"(lds_addr), "v"((uint64_t)(uintptr_t)gptr)
                 : "memory");
}
__device__ __forceinline__ void wait_async0() {
    asm volatile("s_wait_asynccnt 0x0" ::: "memory");
}

// Load one 16x16x32 bf16 WMMA fragment for this lane from a row-major LDS
// tile. Per ISA 7.12.2 (16-bit A/B layout): lanes 0-15 hold K=0..7 and
// K=16..23; lanes 16-31 hold K=8..15 and K=24..31 — i.e. two contiguous
// 8-elem (16B) runs at kb and kb+16 with kb = 8*(lane>>4).
__device__ __forceinline__ v16bf load_frag(const bf16_t* row, int kb) {
    union { v16bf v; uint4 q[2]; } u;
    u.q[0] = *(const uint4*)(row + kb);
    u.q[1] = *(const uint4*)(row + kb + 16);
    return u.v;
}

// ---------------- kernels ----------------

// f32 -> bf16 grid-stride convert
__global__ void cvt_bf16_kernel(const float* __restrict__ src,
                                bf16_t* __restrict__ dst, long n) {
    long i = (long)blockIdx.x * blockDim.x + threadIdx.x;
    long stride = (long)gridDim.x * blockDim.x;
    for (; i < n; i += stride) dst[i] = (bf16_t)src[i];
}

// embedding gather: x[tok] = emb[ids[tok]], f32 + bf16 copies
__global__ __launch_bounds__(256) void embed_kernel(
    const int* __restrict__ ids, const float* __restrict__ emb,
    float* __restrict__ x, bf16_t* __restrict__ xbf) {
    int tok = blockIdx.x;
    int tid = threadIdx.x;
    int id  = ids[tok];
#pragma unroll
    for (int j = 0; j < DIM / 256; ++j) {
        int i = tid + j * 256;
        float v = emb[(size_t)id * DIM + i];
        x[(size_t)tok * DIM + i]   = v;
        xbf[(size_t)tok * DIM + i] = (bf16_t)v;
    }
}

// Tiled WMMA GEMM: C[M,N] (f32) = A[M,K] (bf16, row-major) * B[N,K]^T (bf16).
// Block tile 128x128, 8 waves, each wave owns 32x64 = 2x4 WMMA tiles.
// LDS tiles are double-buffered and filled with global_load_async_to_lds_b128
// (ASYNCcnt) so tile k+1 streams in while tile k is consumed by WMMA.
#define BM 128
#define BN 128
#define BK 32
#define LDST 40   // padded LDS row stride (bf16 elems); 80B keeps 16B align
__global__ __launch_bounds__(256) void gemm_bf16_kernel(
    const bf16_t* __restrict__ A, const bf16_t* __restrict__ B,
    float* __restrict__ C, bf16_t* __restrict__ Cbf,
    const float* __restrict__ bias, int M, int N, int K) {
    __shared__ bf16_t sA[2][BM * LDST];
    __shared__ bf16_t sB[2][BN * LDST];

    const int tid  = threadIdx.x;
    const int bm   = blockIdx.y * BM;
    const int bn   = blockIdx.x * BN;
    const int wave = tid >> 5;
    const int lane = tid & 31;
    const int wm   = wave >> 1;   // 0..3  (M direction, 32 rows each)
    const int wn   = wave & 1;    // 0..1  (N direction, 64 cols each)
    const int lm   = lane & 15;
    const int hi   = lane >> 4;
    const int kb   = hi * 8;

    // staging: both tiles are 128x32 bf16 = 8KB -> 32B (2 x b128) per thread
    const int tr = tid >> 1, tc = (tid & 1) * 16;
    int brow = bn + tr;
    if (brow >= N) brow = N - 1;   // clamp edge (dup rows; stores guarded)
    const bf16_t* aG = A + (size_t)(bm + tr) * K + tc;
    const bf16_t* bG = B + (size_t)brow * K + tc;
    const uint32_t sAoff[2] = {
        (uint32_t)(uintptr_t)(&sA[0][tr * LDST + tc]),
        (uint32_t)(uintptr_t)(&sA[1][tr * LDST + tc])};
    const uint32_t sBoff[2] = {
        (uint32_t)(uintptr_t)(&sB[0][tr * LDST + tc]),
        (uint32_t)(uintptr_t)(&sB[1][tr * LDST + tc])};

    v8f zero = {};
    v8f acc[2][4];
#pragma unroll
    for (int i = 0; i < 2; ++i)
#pragma unroll
        for (int j = 0; j < 4; ++j) acc[i][j] = zero;

    const int ntiles = K / BK;

    // prologue: stage tile 0 into buffer 0
    async_copy_b128(sAoff[0],      aG);
    async_copy_b128(sAoff[0] + 16, aG + 8);
    async_copy_b128(sBoff[0],      bG);
    async_copy_b128(sBoff[0] + 16, bG + 8);
    wait_async0();
    __syncthreads();

    for (int kt = 0; kt < ntiles; ++kt) {
        const int cur = kt & 1, nxt = cur ^ 1;
        if (kt + 1 < ntiles) {
            const bf16_t* an = aG + (kt + 1) * BK;
            const bf16_t* bn_ = bG + (kt + 1) * BK;
            async_copy_b128(sAoff[nxt],      an);
            async_copy_b128(sAoff[nxt] + 16, an + 8);
            async_copy_b128(sBoff[nxt],      bn_);
            async_copy_b128(sBoff[nxt] + 16, bn_ + 8);
            if (kt + 4 < ntiles) {   // L2 prefetch a few tiles ahead
                __builtin_prefetch(aG + (kt + 4) * BK, 0, 1);
                __builtin_prefetch(bG + (kt + 4) * BK, 0, 1);
            }
        }

        v16bf aF[2], bF[4];
#pragma unroll
        for (int s = 0; s < 2; ++s)
            aF[s] = load_frag(&sA[cur][(wm * 32 + s * 16 + lm) * LDST], kb);
#pragma unroll
        for (int s = 0; s < 4; ++s)
            bF[s] = load_frag(&sB[cur][(wn * 64 + s * 16 + lm) * LDST], kb);
#pragma unroll
        for (int i = 0; i < 2; ++i)
#pragma unroll
            for (int j = 0; j < 4; ++j)
                acc[i][j] = __builtin_amdgcn_wmma_f32_16x16x32_bf16(
                    false, aF[i], false, bF[j], (short)0, acc[i][j],
                    false, false);

        wait_async0();
        __syncthreads();
    }

    // store: C/D layout — vgpr r, lanes 0-15: M=r, lanes 16-31: M=8+r
#pragma unroll
    for (int i = 0; i < 2; ++i) {
#pragma unroll
        for (int j = 0; j < 4; ++j) {
            int n = bn + wn * 64 + j * 16 + lm;
            if (n < N) {
                float bv = bias ? bias[n] : 0.0f;
#pragma unroll
                for (int r = 0; r < 8; ++r) {
                    int m = bm + wm * 32 + i * 16 + hi * 8 + r;
                    float val = acc[i][j][r] + bv;
                    C[(size_t)m * N + n] = val;
                    if (Cbf) Cbf[(size_t)m * N + n] = (bf16_t)val;
                }
            }
        }
    }
}

// dt = softplus(zxbcdt[..., -32:] + dt_bias)
__global__ void dt_kernel(const float* __restrict__ zxbcdt,
                          const float* __restrict__ dtb,
                          float* __restrict__ dt) {
    int i   = blockIdx.x * 256 + threadIdx.x;  // NTOK*NHEADS = 65536
    int tok = i >> 5;
    int h   = i & 31;
    float v = zxbcdt[(size_t)tok * D_IN_PROJ + (D_IN_PROJ - NHEADS) + h] + dtb[h];
    dt[i] = (v > 20.0f) ? v : log1pf(__expf(v));
}

// causal depthwise conv (D_CONV=4) + SiLU over xBC slice of zxbcdt
__global__ __launch_bounds__(256) void conv_silu_kernel(
    const float* __restrict__ zxbcdt, const float* __restrict__ cw,
    const float* __restrict__ cb, float* __restrict__ xBC) {
    int b = blockIdx.x / (CONV_DIM / 256);          // 2304/256 = 9 blocks/batch
    int c = (blockIdx.x % (CONV_DIM / 256)) * 256 + threadIdx.x;
    float w0 = cw[c * 4 + 0], w1 = cw[c * 4 + 1];
    float w2 = cw[c * 4 + 2], w3 = cw[c * 4 + 3];
    float bias = cb[c];
    float x0 = 0.f, x1 = 0.f, x2 = 0.f;             // t-3, t-2, t-1
    for (int t = 0; t < SEQ; ++t) {
        size_t tok = (size_t)(b * SEQ + t);
        float xt = zxbcdt[tok * D_IN_PROJ + D_INNER + c];
        float v  = bias + w0 * x0 + w1 * x1 + w2 * x2 + w3 * xt;
        xBC[tok * CONV_DIM + c] = silu_f(v);
        x0 = x1; x1 = x2; x2 = xt;
    }
}

// selective-state scan: one block per (batch, head); 256 thr x 32 state elems.
// thread (p = tid>>2, q = tid&3) owns state[p][q*32 .. q*32+31].
__global__ __launch_bounds__(256) void ssm_scan_kernel(
    const float* __restrict__ dt, const float* __restrict__ xBC,
    const float* __restrict__ A_log, const float* __restrict__ Dsk,
    float* __restrict__ y) {
    int b   = blockIdx.x >> 5;
    int h   = blockIdx.x & 31;
    int tid = threadIdx.x;
    int p   = tid >> 2;
    int q   = tid & 3;
    __shared__ float sB[D_STATE], sC[D_STATE], sX[HEADDIM];

    float hs[32];
#pragma unroll
    for (int j = 0; j < 32; ++j) hs[j] = 0.f;
    float Ah = -__expf(A_log[h]);
    float Dh = Dsk[h];

    for (int t = 0; t < SEQ; ++t) {
        int tok = b * SEQ + t;
        size_t base = (size_t)tok * CONV_DIM;
        if (tid < 128) sB[tid] = xBC[base + D_INNER + tid];
        else           sC[tid - 128] = xBC[base + D_INNER + D_STATE + (tid - 128)];
        if (tid < 64)  sX[tid] = xBC[base + h * HEADDIM + tid];
        __syncthreads();

        float dtv = dt[(size_t)tok * NHEADS + h];
        float dA  = __expf(dtv * Ah);
        float xp  = sX[p];
        float dBx = dtv * xp;
        const float4* B4 = (const float4*)(sB + q * 32);
        const float4* C4 = (const float4*)(sC + q * 32);
        float acc = 0.f;
#pragma unroll
        for (int j4 = 0; j4 < 8; ++j4) {
            float4 bv = B4[j4], cv = C4[j4];
            hs[j4*4+0] = hs[j4*4+0] * dA + dBx * bv.x; acc += hs[j4*4+0] * cv.x;
            hs[j4*4+1] = hs[j4*4+1] * dA + dBx * bv.y; acc += hs[j4*4+1] * cv.y;
            hs[j4*4+2] = hs[j4*4+2] * dA + dBx * bv.z; acc += hs[j4*4+2] * cv.z;
            hs[j4*4+3] = hs[j4*4+3] * dA + dBx * bv.w; acc += hs[j4*4+3] * cv.w;
        }
        // reduce across the 4 lanes covering D_STATE
        acc += __shfl_xor(acc, 1, 32);
        acc += __shfl_xor(acc, 2, 32);
        if (q == 0)
            y[(size_t)tok * D_INNER + h * HEADDIM + p] = acc + Dh * xp;
        __syncthreads();
    }
}

// g = y * silu(z); g *= rsqrt(mean(g^2)+1e-5) * norm_w; write f32 (in-place
// over y) and bf16 for the out_proj GEMM.
__global__ __launch_bounds__(256) void gate_norm_kernel(
    const float* __restrict__ zxbcdt, float* __restrict__ y,
    bf16_t* __restrict__ gbf, const float* __restrict__ nw) {
    int tok = blockIdx.x;
    int tid = threadIdx.x;
    __shared__ float red[256];
    float g[D_INNER / 256];
    float ss = 0.f;
#pragma unroll
    for (int j = 0; j < D_INNER / 256; ++j) {
        int i = tid + j * 256;
        float zv = zxbcdt[(size_t)tok * D_IN_PROJ + i];
        float yv = y[(size_t)tok * D_INNER + i];
        float gv = yv * silu_f(zv);
        g[j] = gv;
        ss += gv * gv;
    }
    red[tid] = ss;
    __syncthreads();
    for (int s = 128; s > 0; s >>= 1) {
        if (tid < s) red[tid] += red[tid + s];
        __syncthreads();
    }
    float scale = rsqrtf(red[0] / (float)D_INNER + 1e-5f);
#pragma unroll
    for (int j = 0; j < D_INNER / 256; ++j) {
        int i = tid + j * 256;
        float o = g[j] * scale * nw[i];
        y[(size_t)tok * D_INNER + i]   = o;
        gbf[(size_t)tok * D_INNER + i] = (bf16_t)o;
    }
}

// ---------------- host launcher ----------------
extern "C" void kernel_launch(void* const* d_in, const int* in_sizes, int n_in,
                              void* d_out, int out_size, void* d_ws, size_t ws_size,
                              hipStream_t stream) {
    const int*   ids      = (const int*)d_in[0];
    const float* emb      = (const float*)d_in[1];
    const float* in_w     = (const float*)d_in[2];
    const float* conv_w   = (const float*)d_in[3];
    const float* conv_b   = (const float*)d_in[4];
    const float* dt_bias  = (const float*)d_in[5];
    const float* A_log    = (const float*)d_in[6];
    const float* D_skip   = (const float*)d_in[7];
    const float* norm_w   = (const float*)d_in[8];
    const float* out_w    = (const float*)d_in[9];
    const float* logits_b = (const float*)d_in[10];
    float* logits = (float*)d_out;

    // carve workspace
    char* ws = (char*)d_ws;
    auto carve = [&](size_t bytes) {
        void* p = (void*)ws;
        ws += (bytes + 255) & ~(size_t)255;
        return p;
    };
    float*  x_f32  = (float*)carve((size_t)NTOK * DIM * 4);
    float*  zxbcdt = (float*)carve((size_t)NTOK * D_IN_PROJ * 4);
    float*  xBC    = (float*)carve((size_t)NTOK * CONV_DIM * 4);
    float*  dt     = (float*)carve((size_t)NTOK * NHEADS * 4);
    float*  ybuf   = (float*)carve((size_t)NTOK * D_INNER * 4);
    bf16_t* x_bf   = (bf16_t*)carve((size_t)NTOK * DIM * 2);
    bf16_t* g_bf   = (bf16_t*)carve((size_t)NTOK * D_INNER * 2);
    bf16_t* w_bf   = (bf16_t*)carve((size_t)VOCAB * DIM * 2); // reused per-GEMM

    // 1) embedding
    embed_kernel<<<NTOK, 256, 0, stream>>>(ids, emb, x_f32, x_bf);

    for (int d = 0; d < DEPTH; ++d) {
        // 2) in_proj GEMM (M=2048, N=4384, K=1024)
        cvt_bf16_kernel<<<2048, 256, 0, stream>>>(
            in_w + (size_t)d * D_IN_PROJ * DIM, w_bf, (long)D_IN_PROJ * DIM);
        gemm_bf16_kernel<<<dim3((D_IN_PROJ + BN - 1) / BN, NTOK / BM), 256, 0, stream>>>(
            x_bf, w_bf, zxbcdt, (bf16_t*)nullptr, (const float*)nullptr,
            NTOK, D_IN_PROJ, DIM);

        // 3) dt = softplus(... + dt_bias)
        dt_kernel<<<(NTOK * NHEADS) / 256, 256, 0, stream>>>(
            zxbcdt, dt_bias + d * NHEADS, dt);

        // 4) causal depthwise conv + silu
        conv_silu_kernel<<<BATCH * (CONV_DIM / 256), 256, 0, stream>>>(
            zxbcdt, conv_w + (size_t)d * CONV_DIM * D_CONV,
            conv_b + (size_t)d * CONV_DIM, xBC);

        // 5) selective scan
        ssm_scan_kernel<<<BATCH * NHEADS, 256, 0, stream>>>(
            dt, xBC, A_log + d * NHEADS, D_skip + d * NHEADS, ybuf);

        // 6) gate + RMSNorm
        gate_norm_kernel<<<NTOK, 256, 0, stream>>>(
            zxbcdt, ybuf, g_bf, norm_w + (size_t)d * D_INNER);

        // 7) out_proj GEMM (M=2048, N=1024, K=2048) -> new x (f32 + bf16)
        cvt_bf16_kernel<<<2048, 256, 0, stream>>>(
            out_w + (size_t)d * DIM * D_INNER, w_bf, (long)DIM * D_INNER);
        gemm_bf16_kernel<<<dim3(DIM / BN, NTOK / BM), 256, 0, stream>>>(
            g_bf, w_bf, x_f32, x_bf, (const float*)nullptr,
            NTOK, DIM, D_INNER);
    }

    // 8) logits = x @ emb^T + b  (M=2048, N=32000, K=1024)
    cvt_bf16_kernel<<<4096, 256, 0, stream>>>(emb, w_bf, (long)VOCAB * DIM);
    gemm_bf16_kernel<<<dim3(VOCAB / BN, NTOK / BM), 256, 0, stream>>>(
        x_bf, w_bf, logits, (bf16_t*)nullptr, logits_b,
        NTOK, VOCAB, DIM);
}
